// FNFModel_90460601188963
// MI455X (gfx1250) — matrix operand order repaired and checked by reading, not compiled, not run on hardware
//
#include <hip/hip_runtime.h>
#include <math.h>

// ---------------------------------------------------------------------------
// FNF model forward for MI455X (gfx1250).
// All FLOP-carrying matmuls run on v_wmma_f32_16x16x32_bf16 with bf16
// operands in memory: producers emit bf16 copies, weights are converted once
// per launch, so the GEMM K-loop is pure {2x global_load_b128 -> v_wmma}
// per fragment with zero conversion ALU. 16x64 tile per wave with A-fragment
// reuse; conv im2col zero-padded to K=352; projection split-K x4.
// rfft/irfft over T=63 are f32 VALU kernels with LDS-staged tiles.
// ---------------------------------------------------------------------------

typedef __attribute__((ext_vector_type(16))) __bf16    v16bf;
typedef __attribute__((ext_vector_type(8)))  float     v8f;
typedef __attribute__((ext_vector_type(4)))  unsigned  u32x4;

// global (address_space(1)) pointer to a 16B chunk: forces global_load_b128
// (flat loads would tick DScnt too and contend for the LDS datapath)
typedef const __attribute__((address_space(1))) u32x4* gq4;

constexpr int Bc = 256, Mv = 21, Lc = 512, Pc = 16, Sc = 8;
constexpr int Dc = 256, Tc = 63, Fc = 32, HIDc = 512, Hc = 96;
constexpr int ROWS = Bc * Tc;                 // 16128
constexpr int KC = 336, KCP = 352;            // conv K and padded K
constexpr int NP = Mv * Hc;                   // 2016 proj cols
constexpr long BTD = (long)Bc * Tc * Dc;      // 4,128,768
constexpr long BFD = (long)Bc * Fc * Dc;      // 2,097,152

__device__ __forceinline__ __bf16 tobf(float f) {
  unsigned u = __builtin_bit_cast(unsigned, f) + 0x8000u;  // round-to-nearest
  return __builtin_bit_cast(__bf16, (unsigned short)(u >> 16));
}

struct Q2 { u32x4 a, b; };   // 32 bytes == one v16bf fragment

// A fragment (16x32 MxK): lane needs bf16 elements {0..7} and {16..23}
// relative to its (row, k-offset) pointer -> two global 16B loads.
__device__ __forceinline__ v16bf load_bf_A(const __bf16* p) {
  Q2 q;
  q.a = *(gq4)(unsigned long long)(p);
  q.b = *(gq4)(unsigned long long)(p + 16);
  return __builtin_bit_cast(v16bf, q);
}

// B fragment (32x16 KxN): lane needs bf16 elements {0..7} and {16..23} -> two global 16B loads.
__device__ __forceinline__ v16bf load_bf_B(const __bf16* p) {
  Q2 q;
  q.a = *(gq4)(unsigned long long)(p);
  q.b = *(gq4)(unsigned long long)(p + 16);
  return __builtin_bit_cast(v16bf, q);
}

// ---------------- fast GEMM: C = A1*W1^T (+signB*A2*W2^T) + bscale*bias -----
// Requires: K%32==0, Mr%16==0, Nc%16==0, unit K-strides, bf16 operands.
// Wave computes 16 rows x 64 cols (4 fragments, A reused 4x). Block = 64x128.
template <bool DUAL, bool NG>
__global__ void __launch_bounds__(256)
k_gemm_fast(const __bf16* __restrict__ A1, long a1rs,
            const __bf16* __restrict__ W1, long w1rs,
            const __bf16* __restrict__ A2, long a2rs,
            const __bf16* __restrict__ W2, long w2rs,
            float signB, const float* __restrict__ bias, float bscale,
            float* __restrict__ C, __bf16* __restrict__ Cbf, long cld,
            int Mr, int Nc, int K, int epi) {
  int lane = threadIdx.x & 31;
  int wave = threadIdx.x >> 5;
  int m0 = blockIdx.y * 64 + (wave >> 1) * 16;
  int n0 = blockIdx.x * 128 + (wave & 1) * 64;
  if (m0 >= Mr || n0 >= Nc) return;          // wave-uniform
  int lr = lane & 15;
  int klA = (lane >> 4) << 3;
  int klB = (lane >> 4) << 3;
  const __bf16* pa1 = A1 + (long)(m0 + lr) * a1rs + klA;
  const __bf16* pa2 = DUAL ? A2 + (long)(m0 + lr) * a2rs + klA : nullptr;
  const __bf16* pb1[4];
  const __bf16* pb2[4];
#pragma unroll
  for (int j = 0; j < 4; ++j) {
    int nj = n0 + j * 16;
    if (NG && nj >= Nc) nj = n0;             // safe clamp; result discarded
    pb1[j] = W1 + (long)(nj + lr) * w1rs + klB;
    if (DUAL) pb2[j] = W2 + (long)(nj + lr) * w2rs + klB;
  }
  v8f acc1[4] = {};
  v8f acc2[4] = {};
  for (int k0 = 0; k0 < K; k0 += 32) {
    __builtin_prefetch(pa1 + 64, 0, 1);      // global_prefetch_b8
    v16bf a1f = load_bf_A(pa1);
    pa1 += 32;
    v16bf a2f;
    if (DUAL) { a2f = load_bf_A(pa2); pa2 += 32; }
#pragma unroll
    for (int j = 0; j < 4; ++j) {
      v16bf b1 = load_bf_B(pb1[j]);
      pb1[j] += 32;
      acc1[j] = __builtin_amdgcn_wmma_f32_16x16x32_bf16(
          false, a1f, false, b1, (short)0, acc1[j], false, false);
      if (DUAL) {
        v16bf b2 = load_bf_B(pb2[j]);
        pb2[j] += 32;
        acc2[j] = __builtin_amdgcn_wmma_f32_16x16x32_bf16(
            false, a2f, false, b2, (short)0, acc2[j], false, false);
      }
    }
  }
  int rbase = m0 + ((lane >> 4) << 3);
#pragma unroll
  for (int j = 0; j < 4; ++j) {
    int col = n0 + j * 16 + lr;
    if (NG && n0 + j * 16 >= Nc) continue;
#pragma unroll
    for (int v = 0; v < 8; ++v) {
      int row = rbase + v;
      float xv = acc1[j][v];
      if (DUAL) xv += signB * acc2[j][v];
      if (bias) xv += bscale * bias[col];
      if (epi == 1) xv = 0.5f * xv * (1.0f + erff(xv * 0.7071067811865476f));
      if (C)   C[(long)row * cld + col] = xv;
      if (Cbf) Cbf[(long)row * cld + col] = tobf(xv);
    }
  }
}

// ---------------- f32 DFT kernels with LDS staging --------------------------
// rfft over T: zr[b,f,d] = sum_t cFt[t,f]*x[b,t,d]; zi with sFt.
// Outputs bf16 (consumed only by the complex-linear WMMA GEMMs).
__global__ void __launch_bounds__(256)
k_rfft(const float* x, const float* cFt, const float* sFt,
       __bf16* zr, __bf16* zi) {
  __shared__ float xs[Tc * Dc];   // 63KB of the 320KB WGP LDS
  int b = blockIdx.x, tid = threadIdx.x;
  const float* xb = x + (long)b * Tc * Dc;
  for (int i = tid; i < Tc * Dc; i += 256) xs[i] = xb[i];
  __syncthreads();
  float ar[Fc] = {};
  float ai[Fc] = {};
  for (int t = 0; t < Tc; ++t) {
    float xv = xs[t * Dc + tid];
    const float* cr = cFt + t * Fc;
    const float* sr = sFt + t * Fc;
#pragma unroll
    for (int f = 0; f < Fc; ++f) {
      ar[f] = fmaf(cr[f], xv, ar[f]);
      ai[f] = fmaf(sr[f], xv, ai[f]);
    }
  }
  long o = (long)b * Fc * Dc + tid;
#pragma unroll
  for (int f = 0; f < Fc; ++f) {
    zr[o + (long)f * Dc] = tobf(ar[f]);
    zi[o + (long)f * Dc] = tobf(ai[f]);
  }
}

// irfft: pt[b,t,d] = sum_f ict[t,f]*yr[b,f,d] + ist[t,f]*yi[b,f,d]  (f32 out)
__global__ void __launch_bounds__(256)
k_irfft(const float* yr, const float* yi, const float* ict, const float* ist,
        float* pt) {
  __shared__ float yrs[Fc * Dc];
  __shared__ float yis[Fc * Dc];
  int b = blockIdx.x, tid = threadIdx.x;
  const float* yrb = yr + (long)b * Fc * Dc;
  const float* yib = yi + (long)b * Fc * Dc;
  for (int i = tid; i < Fc * Dc; i += 256) { yrs[i] = yrb[i]; yis[i] = yib[i]; }
  __syncthreads();
  long o = (long)b * Tc * Dc + tid;
  for (int t = 0; t < Tc; ++t) {
    const float* cr = ict + t * Fc;
    const float* sr = ist + t * Fc;
    float a = 0.0f;
#pragma unroll
    for (int f = 0; f < Fc; ++f)
      a += cr[f] * yrs[f * Dc + tid] + sr[f] * yis[f * Dc + tid];
    pt[o + (long)t * Dc] = a;
  }
}

// ---------------- elementwise / norm kernels --------------------------------
__global__ void k_instnorm(const float* x, float* xn, float* meanv, float* stdv) {
  int row = blockIdx.x, tid = threadIdx.x;
  const float* xr = x + (long)row * Lc;
  float a = xr[tid], b = xr[tid + 256];
  __shared__ float rs[256], rq[256];
  rs[tid] = a + b;
  rq[tid] = a * a + b * b;
  __syncthreads();
  for (int o = 128; o > 0; o >>= 1) {
    if (tid < o) { rs[tid] += rs[tid + o]; rq[tid] += rq[tid + o]; }
    __syncthreads();
  }
  float mu = rs[0] * (1.0f / Lc);
  float var = rq[0] * (1.0f / Lc) - mu * mu;
  float sd = sqrtf(fmaxf(var, 0.0f));
  if (tid == 0) { meanv[row] = mu; stdv[row] = sd; }
  float inv = 1.0f / (sd + 1e-5f);
  xn[(long)row * Lc + tid] = (a - mu) * inv;
  xn[(long)row * Lc + tid + 256] = (b - mu) * inv;
}

// generic f32 -> bf16 conversion (weights)
__global__ void k_cvt(const float* a, __bf16* b, long n) {
  long i = (long)blockIdx.x * 256 + threadIdx.x;
  if (i < n) b[i] = tobf(a[i]);
}

// im2col -> bf16 with row stride KCP=352 and zero-padded tail
__global__ void k_im2col(const float* xn, __bf16* Xc, long total) {
  long i = (long)blockIdx.x * 256 + threadIdx.x;
  if (i >= total) return;
  int c = (int)(i % KCP);
  long r = i / KCP;
  float v = 0.0f;
  if (c < KC) {
    int t = (int)(r % Tc);
    int b = (int)(r / Tc);
    int m = c / Pc, p = c % Pc;
    v = xn[((long)b * Mv + m) * Lc + t * Sc + p];
  }
  Xc[i] = tobf(v);
}

// pad conv_w (256 x 336) into bf16 Wc (256 x 352, zero tail)
__global__ void k_padw(const float* cw, __bf16* Wc, int total) {
  int i = blockIdx.x * 256 + threadIdx.x;
  if (i >= total) return;
  int c = i % KCP;
  int r = i / KCP;
  Wc[i] = tobf((c < KC) ? cw[r * KC + c] : 0.0f);
}

// Positional encoding + DFT / inverse-DFT matrices in [t][f] layout
__global__ void k_init(float* pe, float* cFt, float* sFt, float* ict, float* ist) {
  int i = blockIdx.x * 256 + threadIdx.x;
  const float W0 = 6.283185307179586f / (float)Tc;
  if (i < Tc * Dc) {
    int t = i >> 8, d = i & 255;
    int j2 = d & ~1;
    float div = expf(-(float)j2 * (9.210340371976184f / (float)Dc)); // ln(1e4)
    float arg = (float)t * div;
    pe[i] = (d & 1) ? cosf(arg) : sinf(arg);
  }
  if (i < Tc * Fc) {
    int t = i / Fc, f = i % Fc;
    float w = W0 * (float)(f * t);
    cFt[i] = cosf(w);
    sFt[i] = -sinf(w);
    float c = (f == 0) ? 1.0f : 2.0f;      // irfft doubles bins 1..31
    ict[i] = c * cosf(w) * (1.0f / (float)Tc);
    ist[i] = -c * sinf(w) * (1.0f / (float)Tc);
  }
}

// add pos-enc to conv output; produce f32 + bf16 copies for both branches
__global__ void k_addpe(float* tb, float* sb, __bf16* tbb, __bf16* sbb,
                        const float* pe) {
  long i = (long)blockIdx.x * 256 + threadIdx.x;   // grid covers BTD exactly
  int col = (int)(i & 255);
  int t = (int)((i >> 8) % Tc);
  float v = tb[i] + pe[t * Dc + col];
  tb[i] = v;
  sb[i] = v;
  __bf16 h = tobf(v);
  tbb[i] = h;
  sbb[i] = h;
}

__global__ void k_softshrink(const float* yr, const float* yi,
                             __bf16* zr, __bf16* zi, long n) {
  long i = (long)blockIdx.x * 256 + threadIdx.x;
  if (i >= n) return;
  float a = yr[i], b = yi[i];
  float mag = sqrtf(a * a + b * b);
  float sc = fmaxf(mag - 0.01f, 0.0f) / (mag + 1e-8f);
  sc = (mag > 0.01f) ? sc : 0.0f;
  zr[i] = tobf(a * sc);
  zi[i] = tobf(b * sc);
}

__global__ void k_mul(const float* a, const float* b, __bf16* c, long n) {
  long i = (long)blockIdx.x * 256 + threadIdx.x;
  if (i < n) c[i] = tobf(a[i] * b[i]);
}

// x = LayerNorm(add + x)*g + b, written as f32 (x) and bf16 (xb)
__global__ void k_resln(const float* add, float* x, __bf16* xb,
                        const float* g, const float* b) {
  int row = blockIdx.x, tid = threadIdx.x;
  long off = (long)row * Dc + tid;
  float v = add[off] + x[off];
  __shared__ float rs[256], rq[256];
  rs[tid] = v;
  rq[tid] = v * v;
  __syncthreads();
  for (int o = 128; o > 0; o >>= 1) {
    if (tid < o) { rs[tid] += rs[tid + o]; rq[tid] += rq[tid + o]; }
    __syncthreads();
  }
  float mu = rs[0] * (1.0f / Dc);
  float var = rq[0] * (1.0f / Dc) - mu * mu;
  float inv = rsqrtf(var + 1e-5f);
  float r = (v - mu) * inv * g[tid] + b[tid];
  x[off] = r;
  xb[off] = tobf(r);
}

__global__ void k_gate(const float* t, const float* gw, const float* gb,
                       float* gate, int rows) {
  int row = blockIdx.x * 8 + (threadIdx.x >> 5);
  int lane = threadIdx.x & 31;
  if (row >= rows) return;
  const float* tr = t + (long)row * Dc;
  float s = 0.0f;
  for (int i = lane; i < Dc; i += 32) s += tr[i] * gw[i];
  for (int o = 16; o > 0; o >>= 1) s += __shfl_xor(s, o, 32);
  if (lane == 0) gate[row] = 1.0f / (1.0f + expf(-(s + gb[0])));
}

__global__ void k_feat(const float* gate, const float* t, const float* s,
                       __bf16* f) {
  long i = (long)blockIdx.x * 256 + threadIdx.x;
  float g = gate[i >> 8];
  f[i] = tobf(g * t[i] + (1.0f - g) * s[i]);
}

// reduce 4 split-K partials + bias, then de-normalize into the output
__global__ void k_final_red(const float* p0, const float* p1, const float* p2,
                            const float* p3, const float* bias,
                            const float* meanv, const float* stdv,
                            float* y, long n) {
  long i = (long)blockIdx.x * 256 + threadIdx.x;
  if (i >= n) return;
  int b = (int)(i / NP);
  int rem = (int)(i % NP);
  int m = rem / Hc;
  float v = p0[i] + p1[i] + p2[i] + p3[i] + bias[rem];
  y[i] = v * (stdv[b * Mv + m] + 1e-5f) + meanv[b * Mv + m];
}

// ---------------- fast-GEMM launch helper -----------------------------------
static void launch_gemm(hipStream_t stream,
                        const __bf16* A1, long a1rs, const __bf16* W1, long w1rs,
                        const __bf16* A2, long a2rs, const __bf16* W2, long w2rs,
                        float signB, const float* bias, float bscale,
                        float* C, __bf16* Cbf, long cld,
                        int Mr, int Nc, int K, int epi) {
  dim3 g((Nc + 127) / 128, (Mr + 63) / 64, 1);
  bool ng = (Nc % 128) != 0;
  if (A2) {
    if (ng)
      k_gemm_fast<true, true><<<g, dim3(256), 0, stream>>>(
          A1, a1rs, W1, w1rs, A2, a2rs, W2, w2rs, signB, bias, bscale,
          C, Cbf, cld, Mr, Nc, K, epi);
    else
      k_gemm_fast<true, false><<<g, dim3(256), 0, stream>>>(
          A1, a1rs, W1, w1rs, A2, a2rs, W2, w2rs, signB, bias, bscale,
          C, Cbf, cld, Mr, Nc, K, epi);
  } else {
    if (ng)
      k_gemm_fast<false, true><<<g, dim3(256), 0, stream>>>(
          A1, a1rs, W1, w1rs, A2, a2rs, W2, w2rs, signB, bias, bscale,
          C, Cbf, cld, Mr, Nc, K, epi);
    else
      k_gemm_fast<false, false><<<g, dim3(256), 0, stream>>>(
          A1, a1rs, W1, w1rs, A2, a2rs, W2, w2rs, signB, bias, bscale,
          C, Cbf, cld, Mr, Nc, K, epi);
  }
}

extern "C" void kernel_launch(void* const* d_in, const int* in_sizes, int n_in,
                              void* d_out, int out_size, void* d_ws, size_t ws_size,
                              hipStream_t stream) {
  (void)in_sizes; (void)n_in; (void)out_size; (void)ws_size;
  const float* x      = (const float*)d_in[0];
  const float* conv_w = (const float*)d_in[1];
  const float* conv_b = (const float*)d_in[2];
  const float* gate_w = (const float*)d_in[35];
  const float* gate_b = (const float*)d_in[36];
  const float* proj_w = (const float*)d_in[37];
  const float* proj_b = (const float*)d_in[38];

  constexpr long GEW_SZ = 3L * HIDc * Dc;      // per-branch Gew/Gpw stack
  constexpr long DD_SZ  = 3L * Dc * Dc;        // per-branch DxD stack
  constexpr long WBR_SZ = 2 * GEW_SZ + 5 * DD_SZ;

  // ---- workspace layout (~250 MB) ---------------------------------------
  float* w = (float*)d_ws;
  size_t o = 0;
  auto alloc  = [&](size_t n) { float* p = w + o; o += n; return p; };
  auto allocb = [&](size_t n) { __bf16* p = (__bf16*)(w + o); o += (n + 1) / 2; return p; };
  float*  xn    = alloc((size_t)Bc * Mv * Lc);
  float*  meanv = alloc((size_t)Bc * Mv);
  float*  stdv  = alloc((size_t)Bc * Mv);
  float*  pe    = alloc((size_t)Tc * Dc);
  float*  cFt   = alloc((size_t)Tc * Fc);
  float*  sFt   = alloc((size_t)Tc * Fc);
  float*  ict   = alloc((size_t)Tc * Fc);
  float*  ist   = alloc((size_t)Tc * Fc);
  float*  tbuf  = alloc(BTD);
  float*  sbuf  = alloc(BTD);
  float*  gbuf  = alloc(BTD);
  float*  yr    = alloc(BFD);
  float*  yi    = alloc(BFD);
  float*  ptb   = alloc(BTD);
  float*  blko  = alloc(BTD);
  float*  gate  = alloc((size_t)ROWS);
  float*  pj0   = alloc((size_t)Bc * NP);
  float*  pj1   = alloc((size_t)Bc * NP);
  float*  pj2   = alloc((size_t)Bc * NP);
  float*  pj3   = alloc((size_t)Bc * NP);
  __bf16* Xcb   = allocb((size_t)ROWS * KCP);
  __bf16* Wcb   = allocb((size_t)Dc * KCP);
  __bf16* tb16  = allocb(BTD);
  __bf16* sb16  = allocb(BTD);
  __bf16* h1b   = allocb((size_t)ROWS * HIDc);
  __bf16* zrb   = allocb(BFD);
  __bf16* zib   = allocb(BFD);
  __bf16* gpbb  = allocb(BTD);
  __bf16* featb = allocb(BTD);
  __bf16* wbr   = allocb((size_t)2 * WBR_SZ);
  __bf16* pwb   = allocb((size_t)NP * Tc * Dc);

  // ---- weight conversions (once per launch) ------------------------------
  auto cvt = [&](const float* src, __bf16* dst, long n) {
    k_cvt<<<(unsigned)((n + 255) / 256), 256, 0, stream>>>(src, dst, n);
  };
  for (int br = 0; br < 2; ++br) {
    int pb = 3 + br * 16;
    __bf16* wb = wbr + (long)br * WBR_SZ;
    cvt((const float*)d_in[pb + 2], wb, GEW_SZ);                         // Gew
    cvt((const float*)d_in[pb + 4], wb + GEW_SZ, GEW_SZ);                // Gpw
    cvt((const float*)d_in[pb + 0], wb + 2 * GEW_SZ, DD_SZ);             // Tw
    cvt((const float*)d_in[pb + 6], wb + 2 * GEW_SZ + DD_SZ, DD_SZ);     // H1Wr
    cvt((const float*)d_in[pb + 7], wb + 2 * GEW_SZ + 2 * DD_SZ, DD_SZ); // H1Wi
    cvt((const float*)d_in[pb + 10], wb + 2 * GEW_SZ + 3 * DD_SZ, DD_SZ);// H2Wr
    cvt((const float*)d_in[pb + 11], wb + 2 * GEW_SZ + 4 * DD_SZ, DD_SZ);// H2Wi
  }
  cvt(proj_w, pwb, (long)NP * Tc * Dc);
  k_padw<<<(Dc * KCP + 255) / 256, 256, 0, stream>>>(conv_w, Wcb, Dc * KCP);

  // ---- preprocessing ----------------------------------------------------
  k_instnorm<<<Bc * Mv, 256, 0, stream>>>(x, xn, meanv, stdv);
  k_init<<<(Tc * Dc + 255) / 256, 256, 0, stream>>>(pe, cFt, sFt, ict, ist);
  long nim = (long)ROWS * KCP;
  k_im2col<<<(unsigned)((nim + 255) / 256), 256, 0, stream>>>(xn, Xcb, nim);

  // conv as fast GEMM (K padded to 352) + bias -> tbuf; then +pos-enc, dup
  launch_gemm(stream, Xcb, KCP, Wcb, KCP, nullptr, 0, nullptr, 0,
              0.0f, conv_b, 1.0f, tbuf, nullptr, Dc, ROWS, Dc, KCP, 0);
  k_addpe<<<ROWS, 256, 0, stream>>>(tbuf, sbuf, tb16, sb16, pe);

  // ---- 2 branches x 3 blocks -------------------------------------------
  for (int br = 0; br < 2; ++br) {
    float*  xinb  = br ? sbuf : tbuf;
    __bf16* xinbb = br ? sb16 : tb16;
    int pb = 3 + br * 16;
    __bf16* wb    = wbr + (long)br * WBR_SZ;
    __bf16* Gewb  = wb;
    __bf16* Gpwb  = wb + GEW_SZ;
    __bf16* Twb   = wb + 2 * GEW_SZ;
    __bf16* H1Wrb = wb + 2 * GEW_SZ + DD_SZ;
    __bf16* H1Wib = wb + 2 * GEW_SZ + 2 * DD_SZ;
    __bf16* H2Wrb = wb + 2 * GEW_SZ + 3 * DD_SZ;
    __bf16* H2Wib = wb + 2 * GEW_SZ + 4 * DD_SZ;
    const float* Tb   = (const float*)d_in[pb + 1];
    const float* Geb  = (const float*)d_in[pb + 3];
    const float* Gpb  = (const float*)d_in[pb + 5];
    const float* H1bi = (const float*)d_in[pb + 9];   // H1br cancels exactly
    const float* H2bi = (const float*)d_in[pb + 13];  // H2br cancels exactly
    const float* lng  = (const float*)d_in[pb + 14];
    const float* lnb  = (const float*)d_in[pb + 15];

    for (int i = 0; i < 3; ++i) {
      long dd = (long)Dc * Dc;
      // GELU MLP gate path: g = gelu(x@Gew.T+Geb)@Gpw.T + Gpb
      launch_gemm(stream, xinbb, Dc, Gewb + (long)i * HIDc * Dc, Dc,
                  nullptr, 0, nullptr, 0,
                  0.0f, Geb + (long)i * HIDc, 1.0f,
                  nullptr, h1b, HIDc, ROWS, HIDc, Dc, 1);
      launch_gemm(stream, h1b, HIDc, Gpwb + (long)i * Dc * HIDc, HIDc,
                  nullptr, 0, nullptr, 0,
                  0.0f, Gpb + (long)i * Dc, 1.0f,
                  gbuf, nullptr, Dc, ROWS, Dc, HIDc, 0);
      // rfft over T (f32 math, bf16 outputs for the WMMA GEMMs)
      k_rfft<<<Bc, 256, 0, stream>>>(xinb, cFt, sFt, zrb, zib);
      // complex linear 1: yr = zr@Wr.T - zi@Wi.T ; yi = zr@Wi.T + zi@Wr.T + 2*bi
      launch_gemm(stream, zrb, Dc, H1Wrb + i * dd, Dc, zib, Dc, H1Wib + i * dd, Dc,
                  -1.0f, nullptr, 0.0f, yr, nullptr, Dc, Bc * Fc, Dc, Dc, 0);
      launch_gemm(stream, zrb, Dc, H1Wib + i * dd, Dc, zib, Dc, H1Wrb + i * dd, Dc,
                  1.0f, H1bi + (long)i * Dc, 2.0f, yi, nullptr, Dc, Bc * Fc, Dc, Dc, 0);
      k_softshrink<<<(unsigned)(BFD / 256), 256, 0, stream>>>(yr, yi, zrb, zib, BFD);
      // complex linear 2
      launch_gemm(stream, zrb, Dc, H2Wrb + i * dd, Dc, zib, Dc, H2Wib + i * dd, Dc,
                  -1.0f, nullptr, 0.0f, yr, nullptr, Dc, Bc * Fc, Dc, Dc, 0);
      launch_gemm(stream, zrb, Dc, H2Wib + i * dd, Dc, zib, Dc, H2Wrb + i * dd, Dc,
                  1.0f, H2bi + (long)i * Dc, 2.0f, yi, nullptr, Dc, Bc * Fc, Dc, Dc, 0);
      // irfft (f32, LDS-staged)
      k_irfft<<<Bc, 256, 0, stream>>>(yr, yi, ict, ist, ptb);
      // out = (g * pt) @ Tw.T + Tb, then residual + LayerNorm
      k_mul<<<ROWS, 256, 0, stream>>>(gbuf, ptb, gpbb, BTD);
      launch_gemm(stream, gpbb, Dc, Twb + i * dd, Dc, nullptr, 0, nullptr, 0,
                  0.0f, Tb + (long)i * Dc, 1.0f,
                  blko, nullptr, Dc, ROWS, Dc, Dc, 0);
      k_resln<<<ROWS, 256, 0, stream>>>(blko, xinb, xinbb,
                                        lng + (long)i * Dc, lnb + (long)i * Dc);
    }
  }

  // ---- gate-combine, split-K projection, reduce + rescale ----------------
  k_gate<<<(ROWS + 7) / 8, 256, 0, stream>>>(tbuf, gate_w, gate_b, gate, ROWS);
  k_feat<<<ROWS, 256, 0, stream>>>(gate, tbuf, sbuf, featb);
  float* pj[4] = {pj0, pj1, pj2, pj3};
  constexpr int KSP = (Tc * Dc) / 4;   // 4032 per split, %32 == 0
  for (int kk = 0; kk < 4; ++kk) {
    launch_gemm(stream, featb + (long)kk * KSP, (long)Tc * Dc,
                pwb + (long)kk * KSP, (long)Tc * Dc,
                nullptr, 0, nullptr, 0,
                0.0f, nullptr, 0.0f,
                pj[kk], nullptr, NP, Bc, NP, KSP, 0);
  }
  long nout = (long)Bc * NP;
  k_final_red<<<(unsigned)((nout + 255) / 256), 256, 0, stream>>>(
      pj0, pj1, pj2, pj3, proj_b, meanv, stdv, (float*)d_out, nout);
}
